// Graph_Representation_35433480192603
// MI455X (gfx1250) — matrix operand order repaired and checked by reading, not compiled, not run on hardware
//
#include <hip/hip_runtime.h>

// ---------------- problem constants ----------------
constexpr int kN  = 100000;     // nodes
constexpr int kE  = 1600000;    // edges
constexpr int kD  = 64;         // node dim
constexpr int kDE = 32;         // edge dim
constexpr int kG  = 50;         // graph dim
constexpr int kP  = 2;          // prop steps
constexpr int kF1 = 2 * kD + kDE;   // 160 input features of edge MLP
constexpr int kNT = kN / 16;        // 6250 node tiles

// ---------------- types ----------------
typedef __bf16 bf16_t;
typedef __attribute__((ext_vector_type(16))) __bf16 v16bf;
typedef __attribute__((ext_vector_type(8)))  __bf16 v8bf;
typedef __attribute__((ext_vector_type(8)))  float  v8f;

union V16 { v16bf v; v8bf h[2]; };

__device__ __forceinline__ bf16_t f2bf(float f) {
    unsigned u = __builtin_bit_cast(unsigned, f);
    unsigned r = u + 0x7FFFu + ((u >> 16) & 1u);   // round-to-nearest-even
    unsigned short s = (unsigned short)(r >> 16);
    return __builtin_bit_cast(bf16_t, s);
}

__device__ __forceinline__ v8f wmma_bf16(v16bf a, v16bf b, v8f c) {
    // D = A(16x32 bf16) * B(32x16 bf16) + C(f32)
    return __builtin_amdgcn_wmma_f32_16x16x32_bf16(
        /*neg_a=*/false, a, /*neg_b=*/false, b,
        /*c_mod=*/(short)0, c, /*reuse_a=*/false, /*reuse_b=*/false);
}

__device__ __forceinline__ float sigmoidf_(float x) {
    return 1.0f / (1.0f + __expf(-x));
}
__device__ __forceinline__ float tanhf_(float x) {
    float e = __expf(-2.0f * x);
    return (1.0f - e) / (1.0f + e);
}

__device__ __forceinline__ v8bf pack8(float4 a, float4 b) {
    v8bf r;
    r[0] = f2bf(a.x); r[1] = f2bf(a.y); r[2] = f2bf(a.z); r[3] = f2bf(a.w);
    r[4] = f2bf(b.x); r[5] = f2bf(b.y); r[6] = f2bf(b.z); r[7] = f2bf(b.w);
    return r;
}

// ---------------- init: h = x, c = 0, a = 0, out = 0 ----------------
__global__ void init_kernel(const float* __restrict__ x,
                            float* __restrict__ h, float* __restrict__ a,
                            float* __restrict__ c, float* __restrict__ out,
                            int total) {
    int idx = blockIdx.x * 256 + threadIdx.x;
    if (idx < total) { h[idx] = x[idx]; a[idx] = 0.0f; c[idx] = 0.0f; }
    if (idx < kG) out[idx] = 0.0f;
}

// ---------------- prep: fp32 weights -> bf16 images in ws ----------------
__global__ void prep_weights(const float* __restrict__ w1, const float* __restrict__ w2,
                             const float* __restrict__ wih, const float* __restrict__ whh,
                             bf16_t* __restrict__ w1b, bf16_t* __restrict__ w2b,
                             bf16_t* __restrict__ wihb, bf16_t* __restrict__ whhb) {
    int idx = blockIdx.x * 256 + threadIdx.x;
    if (idx < kP * kD * kF1) w1b[idx]  = f2bf(w1[idx]);
    if (idx < kP * kD * kD)  w2b[idx]  = f2bf(w2[idx]);
    if (idx < kP * 4 * kD * kD) { wihb[idx] = f2bf(wih[idx]); whhb[idx] = f2bf(whh[idx]); }
}

// ---------------- edge message kernel ----------------
// Per wave: one 16-edge tile. Gather [h[dst] | h[src] | edge_attr] -> LDS bf16,
// layer1 (K=160) via 20 WMMA, relu -> LDS, layer2 (K=64) via 8 WMMA,
// scatter-add f32 results into a[dst].
constexpr int WAVES_E = 8;
constexpr int XSTR = 168;   // padded bf16 stride (mult of 8 -> 16B rows, odd bank walk)
constexpr int YSTR = 72;

__global__ __launch_bounds__(256) void edge_kernel(
    const float* __restrict__ h, const float* __restrict__ ea,
    const int* __restrict__ ei,
    const bf16_t* __restrict__ w1b, const float* __restrict__ b1,
    const bf16_t* __restrict__ w2b, const float* __restrict__ b2,
    float* __restrict__ a) {
    __shared__ __align__(16) bf16_t Xs[WAVES_E][16][XSTR];
    __shared__ __align__(16) bf16_t Ys[WAVES_E][16][YSTR];

    const int tid  = threadIdx.x;
    const int w    = tid >> 5;
    const int lane = tid & 31;
    const int m15  = lane & 15;
    const int hf   = lane >> 4;
    const int tile = blockIdx.x * WAVES_E + w;      // exact: E/16 tiles
    const int ebase = tile * 16;

    // ---- stage 16 x 160 bf16 input tile (coalesced float4 gathers) ----
    for (int it = lane; it < 16 * 40; it += 32) {
        int e  = it / 40;
        int f4 = (it % 40) * 4;
        int ge = ebase + e;
        const float* p;
        if (f4 < 64)        p = h  + (size_t)ei[kE + ge] * kD + f4;        // h[dst]
        else if (f4 < 128)  p = h  + (size_t)ei[ge]      * kD + (f4 - 64); // h[src]
        else                p = ea + (size_t)ge * kDE + (f4 - 128);
        float4 v = *(const float4*)p;
        bf16_t* xp = &Xs[w][e][f4];
        xp[0] = f2bf(v.x); xp[1] = f2bf(v.y); xp[2] = f2bf(v.z); xp[3] = f2bf(v.w);
    }
    __builtin_amdgcn_wave_barrier();   // LDS is in-order per wave; pin compiler order

    // ---- layer 1: [16x160] x [160x64] ----
    v8f acc1[4];
    #pragma unroll
    for (int n = 0; n < 4; ++n) {
        float s = b1[n * 16 + m15];
        #pragma unroll
        for (int r = 0; r < 8; ++r) acc1[n][r] = s;
    }
    #pragma unroll
    for (int kb = 0; kb < 5; ++kb) {
        V16 afr;
        afr.h[0] = *(const v8bf*)&Xs[w][m15][kb * 32 + hf * 8];
        afr.h[1] = *(const v8bf*)&Xs[w][m15][kb * 32 + 16 + hf * 8];
        #pragma unroll
        for (int n = 0; n < 4; ++n) {
            V16 bfr;
            const bf16_t* bp = w1b + (size_t)(n * 16 + m15) * kF1 + kb * 32 + hf * 8;
            bfr.h[0] = *(const v8bf*)bp;
            bfr.h[1] = *(const v8bf*)(bp + 16);
            acc1[n] = wmma_bf16(afr.v, bfr.v, acc1[n]);
        }
    }

    // ---- relu -> LDS (A layout for layer 2) ----
    #pragma unroll
    for (int n = 0; n < 4; ++n) {
        #pragma unroll
        for (int r = 0; r < 8; ++r) {
            float y = acc1[n][r];
            Ys[w][hf * 8 + r][n * 16 + m15] = f2bf(y > 0.0f ? y : 0.0f);
        }
    }
    __builtin_amdgcn_wave_barrier();

    // ---- layer 2: [16x64] x [64x64], scatter-add into a[dst] ----
    int dsts[8];
    #pragma unroll
    for (int r = 0; r < 8; ++r) dsts[r] = ei[kE + ebase + hf * 8 + r];

    #pragma unroll
    for (int n = 0; n < 4; ++n) {
        float s = b2[n * 16 + m15];
        v8f acc2;
        #pragma unroll
        for (int r = 0; r < 8; ++r) acc2[r] = s;
        #pragma unroll
        for (int kb = 0; kb < 2; ++kb) {
            V16 afr;
            afr.h[0] = *(const v8bf*)&Ys[w][m15][kb * 32 + hf * 8];
            afr.h[1] = *(const v8bf*)&Ys[w][m15][kb * 32 + 16 + hf * 8];
            V16 bfr;
            const bf16_t* bp = w2b + (size_t)(n * 16 + m15) * kD + kb * 32 + hf * 8;
            bfr.h[0] = *(const v8bf*)bp;
            bfr.h[1] = *(const v8bf*)(bp + 16);
            acc2 = wmma_bf16(afr.v, bfr.v, acc2);
        }
        #pragma unroll
        for (int r = 0; r < 8; ++r)
            atomicAdd(a + (size_t)dsts[r] * kD + n * 16 + m15, acc2[r]);
    }
}

// ---------------- LSTM node kernel ----------------
// Per wave: one 16-node tile. gates = h@Wih^T + a@Whh^T + b  (K=64, Nout=256),
// then elementwise LSTM update of (h, c) in place; re-zeros a for next step.
__global__ __launch_bounds__(256) void lstm_kernel(
    float* __restrict__ h, float* __restrict__ a, float* __restrict__ c,
    const bf16_t* __restrict__ wihb, const bf16_t* __restrict__ whhb,
    const float* __restrict__ bih, const float* __restrict__ bhh,
    int ntiles) {
    const int tid  = threadIdx.x;
    const int w    = tid >> 5;
    const int lane = tid & 31;
    const int m15  = lane & 15;
    const int hf   = lane >> 4;
    const int t    = blockIdx.x * 8 + w;
    if (t >= ntiles) return;           // wave-uniform: EXEC stays all-ones

    const size_t rowbase = (size_t)(t * 16 + m15) * kD;

    V16 Ah[2], Aa[2];
    #pragma unroll
    for (int kb = 0; kb < 2; ++kb) {
        const float* hp = h + rowbase + kb * 32 + hf * 8;
        const float* ap = a + rowbase + kb * 32 + hf * 8;
        float4 h0 = *(const float4*)hp,        h1 = *(const float4*)(hp + 4);
        float4 h2 = *(const float4*)(hp + 16), h3 = *(const float4*)(hp + 20);
        float4 a0 = *(const float4*)ap,        a1 = *(const float4*)(ap + 4);
        float4 a2 = *(const float4*)(ap + 16), a3 = *(const float4*)(ap + 20);
        Ah[kb].h[0] = pack8(h0, h1); Ah[kb].h[1] = pack8(h2, h3);
        Aa[kb].h[0] = pack8(a0, a1); Aa[kb].h[1] = pack8(a2, a3);
    }
    // zero the aggregate rows for the next propagation step (loads consumed above)
    float4 z4; z4.x = z4.y = z4.z = z4.w = 0.0f;
    #pragma unroll
    for (int kb = 0; kb < 2; ++kb) {
        float* ap = a + rowbase + kb * 32 + hf * 8;
        *(float4*)ap = z4; *(float4*)(ap + 4) = z4;
        *(float4*)(ap + 16) = z4; *(float4*)(ap + 20) = z4;
    }

    #pragma unroll
    for (int q = 0; q < 4; ++q) {              // feature column group
        v8f g4[4];                              // i, f, g, o tiles
        #pragma unroll
        for (int j = 0; j < 4; ++j) {           // gate
            const int nb = j * 4 + q;           // output N-block (0..15)
            const int of = nb * 16 + m15;       // output unit (0..255)
            float s = bih[of] + bhh[of];
            v8f acc;
            #pragma unroll
            for (int r = 0; r < 8; ++r) acc[r] = s;
            #pragma unroll
            for (int kb = 0; kb < 2; ++kb) {
                V16 bi, bh;
                const bf16_t* bp = wihb + (size_t)of * kD + kb * 32 + hf * 8;
                bi.h[0] = *(const v8bf*)bp;       bi.h[1] = *(const v8bf*)(bp + 16);
                const bf16_t* bq = whhb + (size_t)of * kD + kb * 32 + hf * 8;
                bh.h[0] = *(const v8bf*)bq;       bh.h[1] = *(const v8bf*)(bq + 16);
                acc = wmma_bf16(Ah[kb].v, bi.v, acc);
                acc = wmma_bf16(Aa[kb].v, bh.v, acc);
            }
            g4[j] = acc;
        }
        const int col = q * 16 + m15;
        #pragma unroll
        for (int r = 0; r < 8; ++r) {
            size_t idx = (size_t)(t * 16 + hf * 8 + r) * kD + col;
            float cold = c[idx];
            float ig = sigmoidf_(g4[0][r]);
            float fg = sigmoidf_(g4[1][r]);
            float gg = tanhf_(g4[2][r]);
            float og = sigmoidf_(g4[3][r]);
            float cn = fg * cold + ig * gg;
            c[idx] = cn;
            h[idx] = og * tanhf_(cn);
        }
    }
}

// ---------------- readout: sum_v sigmoid(h gm^T+gb) * (h fm^T+fb) ----------------
__global__ __launch_bounds__(256) void readout_kernel(
    const float* __restrict__ h,
    const float* __restrict__ gw, const float* __restrict__ gb,
    const float* __restrict__ fw, const float* __restrict__ fb,
    float* __restrict__ out) {
    __shared__ float gws[kG * kD];
    __shared__ float fws[kG * kD];
    __shared__ float gbs[kG], fbs[kG], accs[kG];
    for (int idx = threadIdx.x; idx < kG * kD; idx += 256) { gws[idx] = gw[idx]; fws[idx] = fw[idx]; }
    if (threadIdx.x < kG) { gbs[threadIdx.x] = gb[threadIdx.x]; fbs[threadIdx.x] = fb[threadIdx.x]; accs[threadIdx.x] = 0.0f; }
    __syncthreads();

    const int v = blockIdx.x * 256 + threadIdx.x;
    const bool valid = v < kN;
    float hv[kD];
    if (valid) {
        #pragma unroll
        for (int k = 0; k < kD; k += 4) {
            float4 t = *(const float4*)(h + (size_t)v * kD + k);
            hv[k] = t.x; hv[k + 1] = t.y; hv[k + 2] = t.z; hv[k + 3] = t.w;
        }
    }
    for (int j = 0; j < kG; ++j) {
        float s = 0.0f;
        if (valid) {
            float dg = gbs[j], df = fbs[j];
            #pragma unroll
            for (int k = 0; k < kD; ++k) { dg += hv[k] * gws[j * kD + k]; df += hv[k] * fws[j * kD + k]; }
            s = sigmoidf_(dg) * df;
        }
        #pragma unroll
        for (int off = 16; off > 0; off >>= 1) s += __shfl_xor(s, off, 32);
        if ((threadIdx.x & 31) == 0) atomicAdd(&accs[j], s);
    }
    __syncthreads();
    if (threadIdx.x < kG) atomicAdd(&out[threadIdx.x], accs[threadIdx.x]);
}

// ---------------- host launcher ----------------
extern "C" void kernel_launch(void* const* d_in, const int* in_sizes, int n_in,
                              void* d_out, int out_size, void* d_ws, size_t ws_size,
                              hipStream_t stream) {
    const float* x         = (const float*)d_in[0];
    const float* edge_attr = (const float*)d_in[1];
    const int*   edge_idx  = (const int*)d_in[2];
    const float* fe_w1     = (const float*)d_in[3];
    const float* fe_b1     = (const float*)d_in[4];
    const float* fe_w2     = (const float*)d_in[5];
    const float* fe_b2     = (const float*)d_in[6];
    const float* lstm_wih  = (const float*)d_in[7];
    const float* lstm_whh  = (const float*)d_in[8];
    const float* lstm_bih  = (const float*)d_in[9];
    const float* lstm_bhh  = (const float*)d_in[10];
    const float* gm_w      = (const float*)d_in[11];
    const float* gm_b      = (const float*)d_in[12];
    const float* fm_w      = (const float*)d_in[13];
    const float* fm_b      = (const float*)d_in[14];
    float* out = (float*)d_out;

    // workspace layout (floats then bf16 weight images)
    float* h = (float*)d_ws;
    float* a = h + (size_t)kN * kD;
    float* c = a + (size_t)kN * kD;
    bf16_t* w1b  = (bf16_t*)(c + (size_t)kN * kD);
    bf16_t* w2b  = w1b  + (size_t)kP * kD * kF1;
    bf16_t* wihb = w2b  + (size_t)kP * kD * kD;
    bf16_t* whhb = wihb + (size_t)kP * 4 * kD * kD;

    const int total = kN * kD;
    init_kernel<<<(total + 255) / 256, 256, 0, stream>>>(x, h, a, c, out, total);
    prep_weights<<<(kP * 4 * kD * kD + 255) / 256, 256, 0, stream>>>(
        fe_w1, fe_w2, lstm_wih, lstm_whh, w1b, w2b, wihb, whhb);

    for (int i = 0; i < kP; ++i) {
        edge_kernel<<<kE / (16 * WAVES_E), 256, 0, stream>>>(
            h, edge_attr, edge_idx,
            w1b + (size_t)i * kD * kF1, fe_b1 + i * kD,
            w2b + (size_t)i * kD * kD,  fe_b2 + i * kD, a);
        lstm_kernel<<<(kNT + 7) / 8, 256, 0, stream>>>(
            h, a, c,
            wihb + (size_t)i * 4 * kD * kD, whhb + (size_t)i * 4 * kD * kD,
            lstm_bih + i * 4 * kD, lstm_bhh + i * 4 * kD, kNT);
    }

    readout_kernel<<<(kN + 255) / 256, 256, 0, stream>>>(
        h, gm_w, gm_b, fm_w, fm_b, out);
}
